// EnhancedLFTSformer_43860206027375
// MI455X (gfx1250) — matrix operand order repaired and checked by reading, not compiled, not run on hardware
//
#include <hip/hip_runtime.h>
#include <hip/hip_bf16.h>
#include <math.h>

// ---------------------------------------------------------------------------
// EnhancedLFTSformer forward for MI455X (gfx1250).
//  * all dense GEMMs via v_wmma_f32_16x16x32_f16, f32 accumulate
//  * A pre-converted to f16, B pre-packed f16 (N-major, K-contiguous) so LDS
//    staging is a raw copy -> global_load_async_to_lds_b128 + s_wait_asynccnt
//  * 128x128 block tile, 8 wave32 waves, each 32x64 (8 WMMA / K-step)
// ---------------------------------------------------------------------------

typedef __attribute__((ext_vector_type(16))) _Float16 v16h;
typedef __attribute__((ext_vector_type(8)))  float    v8f;

#define TILE_M 128
#define TILE_N 128
#define TILE_K 32
#define LDSS   40          // f16 elems per LDS row = 80B (16B aligned, 20-bank step)
#define USE_ASYNC_LDS 1    // stage tiles with global_load_async_to_lds_b128

__device__ __forceinline__ void async_b128(unsigned lds_off, const _Float16* gptr)
{
#if USE_ASYNC_LDS
    asm volatile("global_load_async_to_lds_b128 %0, %1, off"
                 :: "v"(lds_off), "v"(gptr) : "memory");
#endif
}

__device__ __forceinline__ void wait_async0()
{
#if USE_ASYNC_LDS
#if __has_builtin(__builtin_amdgcn_s_wait_asynccnt)
    __builtin_amdgcn_s_wait_asynccnt(0);
#else
    asm volatile("s_wait_asynccnt 0x0" ::: "memory");
#endif
#endif
}

// flags: bit0 = relu, bit1 = accumulate into C
// A: f16 row-major (M x K).  Bt: f16 packed (N x K), K contiguous.
__global__ void __launch_bounds__(256)
gemm_wmma_f16(const _Float16* __restrict__ A, const _Float16* __restrict__ Bt,
              const float* __restrict__ bias, const float* __restrict__ resid,
              float* __restrict__ C, const _Float16* __restrict__ Zpad,
              int M, int N, int K, int flags,
              int conv_kk, int T_out, int L_in)
{
    __shared__ _Float16 As[TILE_M * LDSS];
    __shared__ _Float16 Bs[TILE_N * LDSS];

    const int tid  = threadIdx.x;
    const int lane = tid & 31;
    const int wave = tid >> 5;
    const int wr   = wave & 3;     // 4 row blocks of 32
    const int wc   = wave >> 2;    // 2 col blocks of 64
    const int m0   = blockIdx.y * TILE_M;
    const int n0   = blockIdx.x * TILE_N;

    // ---- hoisted staging addresses: thread covers rows tid/4 and tid/4+64,
    //      16-byte chunk (tid&3) of each 64-byte (32-half) tile row ----
    const int srow  = tid >> 2;
    const int schnk = (tid & 3) * 8;          // halfs
    const _Float16* agp[2];
    bool aok[2];
#pragma unroll
    for (int i = 0; i < 2; ++i) {
        int  m    = m0 + srow + 64 * i;
        long arow = -1;
        if (conv_kk >= 0) {
            int bb = m / T_out, t = m - bb * T_out;
            int src = 2 * t + conv_kk - 1;
            if (src >= 0 && src < L_in) arow = (long)bb * L_in + src;
        } else {
            arow = m;
        }
        aok[i] = (arow >= 0);
        agp[i] = aok[i] ? (A + arow * (long)K + schnk) : Zpad;
    }
    const _Float16* bgp[2];
#pragma unroll
    for (int i = 0; i < 2; ++i)
        bgp[i] = Bt + (long)(n0 + srow + 64 * i) * K + schnk;

    unsigned alds[2], blds[2];
#pragma unroll
    for (int i = 0; i < 2; ++i) {
        alds[i] = (unsigned)(uintptr_t)&As[(srow + 64 * i) * LDSS + schnk];
        blds[i] = (unsigned)(uintptr_t)&Bs[(srow + 64 * i) * LDSS + schnk];
    }

    v8f acc[2][4] = {};

    for (int k0 = 0; k0 < K; k0 += TILE_K) {
        // ---- stage tiles (raw f16 copies) ----
#if USE_ASYNC_LDS
#pragma unroll
        for (int i = 0; i < 2; ++i) {
            async_b128(alds[i], aok[i] ? agp[i] + k0 : agp[i]);
            async_b128(blds[i], bgp[i] + k0);
        }
        wait_async0();
#else
#pragma unroll
        for (int i = 0; i < 2; ++i) {
            int4 av = aok[i] ? *(const int4*)(agp[i] + k0) : *(const int4*)agp[i];
            int4 bv = *(const int4*)(bgp[i] + k0);
            *(int4*)&As[(srow + 64 * i) * LDSS + schnk] = av;
            *(int4*)&Bs[(srow + 64 * i) * LDSS + schnk] = bv;
        }
#endif
        __syncthreads();

        // ---- fragments: contiguous k = 16*hh + e in both A and B (same order) ----
        v16h afrag[2], bfrag[4];
        const int rl = lane & 15;
        const int hh = lane >> 4;
#pragma unroll
        for (int s = 0; s < 2; ++s) {
            const _Float16* ap = &As[(wr * 32 + s * 16 + rl) * LDSS];
#pragma unroll
            for (int e = 0; e < 8; ++e) afrag[s][e]     = ap[16 * hh + e];
#pragma unroll
            for (int e = 0; e < 8; ++e) afrag[s][8 + e] = ap[16 * hh + 8 + e];
        }
#pragma unroll
        for (int t = 0; t < 4; ++t) {
            const _Float16* bp = &Bs[(wc * 64 + t * 16 + rl) * LDSS + 16 * hh];
#pragma unroll
            for (int e = 0; e < 16; ++e) bfrag[t][e] = bp[e];
        }
#pragma unroll
        for (int s = 0; s < 2; ++s)
#pragma unroll
            for (int t = 0; t < 4; ++t)
                acc[s][t] = __builtin_amdgcn_wmma_f32_16x16x32_f16(
                    false, afrag[s], false, bfrag[t], (short)0, acc[s][t], false, false);
        __syncthreads();
    }

    // ---- epilogue: bias / accumulate / residual / relu ----
    const bool relu  = (flags & 1) != 0;
    const bool accum = (flags & 2) != 0;
#pragma unroll
    for (int s = 0; s < 2; ++s) {
#pragma unroll
        for (int t = 0; t < 4; ++t) {
            int cb = n0 + wc * 64 + t * 16 + (lane & 15);
            if (cb >= N) continue;
            float bia = bias ? bias[cb] : 0.f;
#pragma unroll
            for (int v = 0; v < 8; ++v) {
                int r = m0 + wr * 32 + s * 16 + v + 8 * (lane >> 4);
                if (r >= M) continue;
                long off = (long)r * N + cb;
                float val = acc[s][t][v] + bia;
                if (accum) val += C[off];
                if (resid) val += resid[off];
                if (relu && val < 0.f) val = 0.f;
                C[off] = val;
            }
        }
    }
}

// ---------------- f32 -> f16 activation conversion (vectorized) ----------------
__global__ void __launch_bounds__(256)
cvt_f16_k(const float* __restrict__ in, _Float16* __restrict__ out, long n)
{
    long i = ((long)blockIdx.x * 256 + threadIdx.x) * 8;
    if (i >= n) return;
    float4 a = *(const float4*)(in + i);
    float4 b = *(const float4*)(in + i + 4);
    _Float16 o[8];
    o[0] = (_Float16)a.x; o[1] = (_Float16)a.y; o[2] = (_Float16)a.z; o[3] = (_Float16)a.w;
    o[4] = (_Float16)b.x; o[5] = (_Float16)b.y; o[6] = (_Float16)b.z; o[7] = (_Float16)b.w;
    *(int4*)(out + i) = *(const int4*)o;
}

// ---------------- weight pack: W(k,n strided) -> Bt[n*K + k] f16 ----------------
__global__ void __launch_bounds__(256)
pack_b_k(const float* __restrict__ W, _Float16* __restrict__ Bt,
         int N, int K, int ldbn, int ldbk, int boff)
{
    long i = (long)blockIdx.x * 256 + threadIdx.x;
    if (i >= (long)N * K) return;
    int k = (int)(i % K);
    int n = (int)(i / K);
    Bt[i] = (_Float16)W[(long)boff + (long)n * ldbn + (long)k * ldbk];
}

// ---------------- zero the 64B OOB pad ----------------
__global__ void zeropad_k(_Float16* z)
{
    if (threadIdx.x < 32) z[threadIdx.x] = (_Float16)0.f;
}

// ---------------- positional encoding add ----------------
__global__ void __launch_bounds__(256)
posenc_k(float* __restrict__ X, long total)
{
    long i = (long)blockIdx.x * 256 + threadIdx.x;
    if (i >= total) return;
    int  c   = (int)(i % 512);
    long row = i / 512;
    int  l   = (int)(row % 4096);
    int  p   = c >> 1;
    float ang = (float)l * __expf((float)(2 * p) * (-9.210340371976184f / 512.f));
    X[i] += (c & 1) ? __cosf(ang) : __sinf(ang);
}

// ---------------- ProbSparse M-score ----------------
__global__ void __launch_bounds__(256)
mscore_k(const float* __restrict__ Q, const float* __restrict__ Kb,
         const int* __restrict__ sidx, float* __restrict__ Mv, int Lx, int S)
{
    long id = (long)blockIdx.x * 256 + threadIdx.x;
    long total = 64L * Lx;
    if (id >= total) return;
    int  l  = (int)(id % Lx);
    int  bh = (int)(id / Lx);
    int  h  = bh & 7, b = bh >> 3;
    const float* q = Q + ((long)b * Lx + l) * 512 + h * 64;
    float qv[64];
#pragma unroll
    for (int j = 0; j < 64; ++j) qv[j] = q[j];
    float mx = -1e30f, sm = 0.f;
    for (int s = 0; s < S; ++s) {
        int kr = sidx[(long)l * S + s];
        const float* kp = Kb + ((long)b * Lx + kr) * 512 + h * 64;
        float d = 0.f;
#pragma unroll
        for (int j = 0; j < 64; ++j) d += qv[j] * kp[j];
        mx = fmaxf(mx, d); sm += d;
    }
    Mv[(long)bh * Lx + l] = mx - sm / (float)S;
}

// ---------------- top-k (iterated argmax in LDS) ----------------
__global__ void __launch_bounds__(256)
topk_k(const float* __restrict__ Mv, int* __restrict__ topi, int Lx, int ntop)
{
    extern __shared__ float sm[];
    float* vals = sm;
    float* rmax = sm + Lx;
    int*   ridx = (int*)(rmax + 256);
    int bh = blockIdx.x;
    const float* src = Mv + (long)bh * Lx;
    for (int i = threadIdx.x; i < Lx; i += 256) vals[i] = src[i];
    __syncthreads();
    for (int t = 0; t < ntop; ++t) {
        float m = -1e30f; int mi = 0;
        for (int i = threadIdx.x; i < Lx; i += 256) {
            float v = vals[i];
            if (v > m) { m = v; mi = i; }
        }
        rmax[threadIdx.x] = m; ridx[threadIdx.x] = mi;
        __syncthreads();
        for (int s = 128; s > 0; s >>= 1) {
            if (threadIdx.x < s && rmax[threadIdx.x + s] > rmax[threadIdx.x]) {
                rmax[threadIdx.x] = rmax[threadIdx.x + s];
                ridx[threadIdx.x] = ridx[threadIdx.x + s];
            }
            __syncthreads();
        }
        if (threadIdx.x == 0) {
            topi[(long)bh * ntop + t] = ridx[0];
            vals[ridx[0]] = -1e30f;
        }
        __syncthreads();
    }
}

// ---------------- reduced-Q scores ----------------
__global__ void __launch_bounds__(256)
scores_k(const float* __restrict__ Q, const float* __restrict__ Kb,
         const int* __restrict__ topi, float* __restrict__ Sc, int Lx, int ntop)
{
    int u = blockIdx.x % ntop, bh = blockIdx.x / ntop;
    int h = bh & 7, b = bh >> 3;
    __shared__ float qs[64];
    if (threadIdx.x < 64) {
        int qr = topi[(long)bh * ntop + u];
        qs[threadIdx.x] = Q[((long)b * Lx + qr) * 512 + h * 64 + threadIdx.x];
    }
    __syncthreads();
    for (int l = threadIdx.x; l < Lx; l += 256) {
        const float* kp = Kb + ((long)b * Lx + l) * 512 + h * 64;
        float d = 0.f;
#pragma unroll 16
        for (int j = 0; j < 64; ++j) d += qs[j] * kp[j];
        Sc[(long)blockIdx.x * Lx + l] = d * 0.125f;
    }
}

// ---------------- row softmax ----------------
__global__ void __launch_bounds__(256)
softmax_k(float* __restrict__ Sc, int Lx)
{
    __shared__ float red[256];
    float* p = Sc + (long)blockIdx.x * Lx;
    int tid = threadIdx.x;
    float m = -1e30f;
    for (int i = tid; i < Lx; i += 256) m = fmaxf(m, p[i]);
    red[tid] = m; __syncthreads();
    for (int s = 128; s > 0; s >>= 1) {
        if (tid < s) red[tid] = fmaxf(red[tid], red[tid + s]);
        __syncthreads();
    }
    float mx = red[0]; __syncthreads();
    float sum = 0.f;
    for (int i = tid; i < Lx; i += 256) { float e = __expf(p[i] - mx); p[i] = e; sum += e; }
    red[tid] = sum; __syncthreads();
    for (int s = 128; s > 0; s >>= 1) {
        if (tid < s) red[tid] += red[tid + s];
        __syncthreads();
    }
    float inv = 1.f / red[0];
    for (int i = tid; i < Lx; i += 256) p[i] *= inv;
}

// ---------------- ctx_top = attn @ V ----------------
__global__ void __launch_bounds__(256)
ctx_k(const float* __restrict__ Sc, const float* __restrict__ V,
      float* __restrict__ ctx_top, int Lx, int ntop)
{
    int bh = blockIdx.x / ntop;
    int h = bh & 7, b = bh >> 3;
    int d = threadIdx.x & 63, ch = threadIdx.x >> 6;
    const float* attn = Sc + (long)blockIdx.x * Lx;
    float acc = 0.f;
    for (int l = ch; l < Lx; l += 4)
        acc += attn[l] * V[((long)b * Lx + l) * 512 + h * 64 + d];
    __shared__ float red[256];
    red[threadIdx.x] = acc; __syncthreads();
    if (ch == 0)
        ctx_top[(long)blockIdx.x * 64 + d] = red[d] + red[64 + d] + red[128 + d] + red[192 + d];
}

// ---------------- vmean over sequence ----------------
__global__ void __launch_bounds__(256)
vmean_k(const float* __restrict__ V, float* __restrict__ vm, int Lx)
{
    int bh = blockIdx.x, h = bh & 7, b = bh >> 3;
    int d = threadIdx.x & 63, ch = threadIdx.x >> 6;
    float acc = 0.f;
    for (int l = ch; l < Lx; l += 4)
        acc += V[((long)b * Lx + l) * 512 + h * 64 + d];
    __shared__ float red[256];
    red[threadIdx.x] = acc; __syncthreads();
    if (ch == 0)
        vm[bh * 64 + d] = (red[d] + red[64 + d] + red[128 + d] + red[192 + d]) / (float)Lx;
}

// ---------------- broadcast vmean into context ----------------
__global__ void __launch_bounds__(256)
assemble_k(const float* __restrict__ vm, float* __restrict__ CTX, long total, int Lx)
{
    long i = (long)blockIdx.x * 256 + threadIdx.x;
    if (i >= total) return;
    int  c   = (int)(i % 512);
    long row = i / 512;
    int  b   = (int)(row / Lx);
    CTX[i] = vm[(b * 8 + (c >> 6)) * 64 + (c & 63)];
}

// ---------------- scatter ctx_top into selected rows ----------------
__global__ void __launch_bounds__(256)
scatter_k(const float* __restrict__ ctx_top, const int* __restrict__ topi,
          float* __restrict__ CTX, int Lx, int ntop)
{
    long i = (long)blockIdx.x * 256 + threadIdx.x;
    long total = 64L * ntop * 64;
    if (i >= total) return;
    int  d  = (int)(i & 63);
    long r  = i >> 6;
    int  u  = (int)(r % ntop);
    int  bh = (int)(r / ntop);
    int  h  = bh & 7, b = bh >> 3;
    int  row = topi[(long)bh * ntop + u];
    CTX[((long)b * Lx + row) * 512 + h * 64 + d] = ctx_top[r * 64 + d];
}

// ---------------- LayerNorm (wave per row, wave32 shfl reduce) ----------------
__global__ void __launch_bounds__(256)
layernorm_k(const float* __restrict__ in, float* __restrict__ out,
            const float* __restrict__ g, const float* __restrict__ b, int M)
{
    int row  = blockIdx.x * 8 + (threadIdx.x >> 5);
    int lane = threadIdx.x & 31;
    if (row >= M) return;
    const float* x = in + (long)row * 512;
    float v[16], s = 0.f, ss = 0.f;
#pragma unroll
    for (int i = 0; i < 16; ++i) {
        float t = x[lane + i * 32];
        v[i] = t; s += t; ss += t * t;
    }
#pragma unroll
    for (int m = 16; m >= 1; m >>= 1) {
        s  += __shfl_xor(s,  m, 32);
        ss += __shfl_xor(ss, m, 32);
    }
    float mean = s * (1.f / 512.f);
    float var  = ss * (1.f / 512.f) - mean * mean;
    float inv  = rsqrtf(var + 1e-5f);
    float* o = out + (long)row * 512;
#pragma unroll
    for (int i = 0; i < 16; ++i) {
        int c = lane + i * 32;
        o[c] = (v[i] - mean) * inv * g[c] + b[c];
    }
}

// ---------------- multiscale pool + fuse ----------------
__global__ void __launch_bounds__(256)
fuse_k(const float* __restrict__ lo, const float* __restrict__ ho,
       const float* __restrict__ qo, float* __restrict__ F, long total)
{
    long i = (long)blockIdx.x * 256 + threadIdx.x;
    if (i >= total) return;
    int  c  = (int)(i % 512);
    long rt = i / 512;
    int  t  = (int)(rt % 1024);
    int  b  = (int)(rt / 1024);
    float a = 0.f;
#pragma unroll
    for (int j = 0; j < 4; ++j) a += lo[((long)b * 4096 + t * 4 + j) * 512 + c];
    a *= 0.25f;
    float h2 = 0.5f * (ho[((long)b * 2048 + t * 2) * 512 + c] +
                       ho[((long)b * 2048 + t * 2 + 1) * 512 + c]);
    F[i] = a + h2 + qo[((long)b * 1024 + t) * 512 + c];
}

// ---------------- mean over time ----------------
__global__ void __launch_bounds__(256)
meantime_k(const float* __restrict__ F2, float* __restrict__ P)
{
    int i = blockIdx.x * 256 + threadIdx.x;
    if (i >= 8 * 512) return;
    int c = i % 512, b = i / 512;
    float s = 0.f;
    for (int t = 0; t < 1024; ++t) s += F2[((long)b * 1024 + t) * 512 + c];
    P[i] = s * (1.f / 1024.f);
}

// ---------------- output head ----------------
__global__ void __launch_bounds__(256)
final_k(const float* __restrict__ P, const float* __restrict__ W,
        const float* __restrict__ bia, float* __restrict__ out)
{
    int i = blockIdx.x * 256 + threadIdx.x;
    if (i >= 8 * 96) return;
    int p = i % 96, b = i / 96;
    float s = bia[p];
    for (int c = 0; c < 512; ++c) s += P[b * 512 + c] * W[c * 96 + p];
    out[i] = s;
}

// ===========================================================================
// Host orchestration
// ===========================================================================
struct GemmCtx {
    _Float16* AH;   // f16 activation arena
    _Float16* BT;   // f16 packed-weight arena
    _Float16* ZP;   // 64B zero pad
    hipStream_t st;
};

static inline void cvtA(const GemmCtx& g, const float* A, long n)
{
    cvt_f16_k<<<dim3((unsigned)((n / 8 + 255) / 256)), 256, 0, g.st>>>(A, g.AH, n);
}

static inline void launch_gemm(const GemmCtx& g, const float* Bw,
                               const float* bias, const float* resid, float* C,
                               int M, int N, int K, int ldbn, int ldbk, int boff,
                               int flags, int ckk, int Tout, int Lin)
{
    long nk = (long)N * K;
    pack_b_k<<<dim3((unsigned)((nk + 255) / 256)), 256, 0, g.st>>>(Bw, g.BT, N, K, ldbn, ldbk, boff);
    dim3 grid((N + TILE_N - 1) / TILE_N, (M + TILE_M - 1) / TILE_M);
    gemm_wmma_f16<<<grid, dim3(256), 0, g.st>>>(g.AH, g.BT, bias, resid, C, g.ZP,
                                                M, N, K, flags, ckk, Tout, Lin);
}

struct LayerP { const float* p[16]; }; // wq wk wv wo bq bk bv bo ln1g ln1b w1 b1 w2 b2 ln2g ln2b

struct WSP {
    float *TMP, *Q, *K, *V, *CTX, *FFNH, *MS, *SC, *CTT, *VM;
    int* TOPI;
};

static void run_layer(const GemmCtx& g, float* X, int Lx, int ntop,
                      const int* sidx, const LayerP& L, const WSP& w)
{
    hipStream_t st = g.st;
    const int M = 8 * Lx;
    const int S = ntop; // sample count == n_top per stage
    // QKV projections: convert X once, three WMMA GEMMs
    cvtA(g, X, (long)M * 512);
    launch_gemm(g, L.p[0], L.p[4], nullptr, w.Q, M, 512, 512, 1, 512, 0, 0, -1, 0, 0);
    launch_gemm(g, L.p[1], L.p[5], nullptr, w.K, M, 512, 512, 1, 512, 0, 0, -1, 0, 0);
    launch_gemm(g, L.p[2], L.p[6], nullptr, w.V, M, 512, 512, 1, 512, 0, 0, -1, 0, 0);
    // sparsity measurement + top-k
    {
        long tot = 64L * Lx;
        mscore_k<<<dim3((unsigned)((tot + 255) / 256)), 256, 0, st>>>(w.Q, w.K, sidx, w.MS, Lx, S);
        size_t shmem = (size_t)(Lx + 512) * sizeof(float);
        topk_k<<<64, 256, shmem, st>>>(w.MS, w.TOPI, Lx, ntop);
    }
    // reduced-Q attention
    scores_k<<<64 * ntop, 256, 0, st>>>(w.Q, w.K, w.TOPI, w.SC, Lx, ntop);
    softmax_k<<<64 * ntop, 256, 0, st>>>(w.SC, Lx);
    ctx_k<<<64 * ntop, 256, 0, st>>>(w.SC, w.V, w.CTT, Lx, ntop);
    vmean_k<<<64, 256, 0, st>>>(w.V, w.VM, Lx);
    {
        long tot = 8L * Lx * 512;
        assemble_k<<<dim3((unsigned)((tot + 255) / 256)), 256, 0, st>>>(w.VM, w.CTX, tot, Lx);
        long tot2 = 64L * ntop * 64;
        scatter_k<<<dim3((unsigned)((tot2 + 255) / 256)), 256, 0, st>>>(w.CTT, w.TOPI, w.CTX, Lx, ntop);
    }
    // output projection + residual, LN1
    cvtA(g, w.CTX, (long)M * 512);
    launch_gemm(g, L.p[3], L.p[7], X, w.TMP, M, 512, 512, 1, 512, 0, 0, -1, 0, 0);
    layernorm_k<<<M / 8, 256, 0, st>>>(w.TMP, X, L.p[8], L.p[9], M);
    // FFN (relu GEMM + proj with residual), LN2.  FFNH aliases Q/K/V/CTX.
    cvtA(g, X, (long)M * 512);
    launch_gemm(g, L.p[10], L.p[11], nullptr, w.FFNH, M, 2048, 512, 1, 2048, 0, 1, -1, 0, 0);
    cvtA(g, w.FFNH, (long)M * 2048);
    launch_gemm(g, L.p[12], L.p[13], X, w.TMP, M, 512, 2048, 1, 512, 0, 0, -1, 0, 0);
    layernorm_k<<<M / 8, 256, 0, st>>>(w.TMP, X, L.p[14], L.p[15], M);
}

extern "C" void kernel_launch(void* const* d_in, const int* in_sizes, int n_in,
                              void* d_out, int out_size, void* d_ws, size_t ws_size,
                              hipStream_t stream)
{
    // ---- inputs (setup_inputs() insertion order, recursively flattened) ----
    auto F = [&](int i) { return (const float*)d_in[i]; };
    const float* x       = F(0);
    const float* proj_w  = F(1);  const float* proj_b  = F(2);
    const float* conv1_w = F(3);  const float* conv1_b = F(4);
    auto layer_at = [&](int base) {
        LayerP L;
        for (int j = 0; j < 16; ++j) L.p[j] = F(base + j);
        return L;
    };
    LayerP Ll[3] = { layer_at(5), layer_at(21), layer_at(37) };
    const float* down1_w = F(53); const float* down1_b = F(54);
    LayerP Lh[2] = { layer_at(55), layer_at(71) };
    const float* down2_w = F(87); const float* down2_b = F(88);
    LayerP Lq[1] = { layer_at(89) };
    const float* fus_w = F(105); const float* fus_b = F(106);
    const float* out_w = F(107); const float* out_b = F(108);
    const int* sidx_l = (const int*)d_in[109];
    const int* sidx_h = (const int*)d_in[110];
    const int* sidx_q = (const int*)d_in[111];

    // ---- workspace layout (float units) ----
    const size_t M1 = 32768, M2 = 16384, M3 = 8192;
    float* ws = (float*)d_ws;
    size_t o = 0;
    auto take = [&](size_t n) { float* p = ws + o; o += n; return p; };
    float* XL    = take(M1 * 512);     // stage-1 activations (ends as l_out)
    float* XH    = take(M2 * 512);     // stage-2 (ends as h_out)
    float* XQ    = take(M3 * 512);     // stage-3 (ends as q_out)
    float* TMP   = take(M1 * 512);
    float* Qb    = take(M1 * 512);
    float* Kb    = take(M1 * 512);
    float* Vb    = take(M1 * 512);
    float* CTX   = take(M1 * 512);
    float* FFNH  = Qb;                 // M1*2048 aliases Q+K+V+CTX exactly
    float* MS    = take(64 * 4096);
    float* SC    = take(64L * 48 * 4096);
    float* CTT   = take(64 * 48 * 64);
    float* VM    = take(64 * 64);
    float* FUSED = take(M3 * 512);
    float* POOLED = take(8 * 512);
    _Float16* AH = (_Float16*)take(M1 * 2048 / 2);   // f16 activation arena
    _Float16* BT = (_Float16*)take(2048 * 512 / 2);  // f16 packed weights
    _Float16* ZP = (_Float16*)take(16);              // 64B zero pad
    int* TOPI    = (int*)(ws + o);

    GemmCtx g{AH, BT, ZP, stream};
    WSP W{TMP, Qb, Kb, Vb, CTX, FFNH, MS, SC, CTT, VM, TOPI};

    zeropad_k<<<1, 32, 0, stream>>>(ZP);

    // ---- embedding: x @ proj_w + b, @ conv1_w + b, + positional encoding ----
    cvtA(g, x, (long)M1 * 32);
    launch_gemm(g, proj_w,  proj_b,  nullptr, TMP, (int)M1, 512, 32,  1, 512, 0, 0, -1, 0, 0);
    cvtA(g, TMP, (long)M1 * 512);
    launch_gemm(g, conv1_w, conv1_b, nullptr, XL,  (int)M1, 512, 512, 1, 512, 0, 0, -1, 0, 0);
    {
        long tot = (long)M1 * 512;
        posenc_k<<<dim3((unsigned)((tot + 255) / 256)), 256, 0, stream>>>(XL, tot);
    }

    // ---- stage L (3 layers @ 4096, n_top = 45) ----
    for (int i = 0; i < 3; ++i) run_layer(g, XL, 4096, 45, sidx_l, Ll[i], W);

    // ---- conv down 1: stride-2 k=3 as 3 accumulating WMMA passes ----
    cvtA(g, XL, (long)M1 * 512);
    for (int kk = 0; kk < 3; ++kk)
        launch_gemm(g, down1_w, kk == 0 ? down1_b : nullptr, nullptr, XH,
                    (int)M2, 512, 512, 512 * 3, 3, kk, kk == 0 ? 0 : 2, kk, 2048, 4096);

    // ---- stage H (2 layers @ 2048, n_top = 40) ----
    for (int i = 0; i < 2; ++i) run_layer(g, XH, 2048, 40, sidx_h, Lh[i], W);

    // ---- conv down 2 ----
    cvtA(g, XH, (long)M2 * 512);
    for (int kk = 0; kk < 3; ++kk)
        launch_gemm(g, down2_w, kk == 0 ? down2_b : nullptr, nullptr, XQ,
                    (int)M3, 512, 512, 512 * 3, 3, kk, kk == 0 ? 0 : 2, kk, 1024, 2048);

    // ---- stage Q (1 layer @ 1024, n_top = 35) ----
    run_layer(g, XQ, 1024, 35, sidx_q, Lq[0], W);

    // ---- fuse multiscale outputs, project, pool, head ----
    {
        long tot = 8L * 1024 * 512;
        fuse_k<<<dim3((unsigned)((tot + 255) / 256)), 256, 0, stream>>>(XL, XH, XQ, FUSED, tot);
    }
    cvtA(g, FUSED, (long)M3 * 512);
    launch_gemm(g, fus_w, fus_b, nullptr, TMP, (int)M3, 512, 512, 1, 512, 0, 0, -1, 0, 0);
    meantime_k<<<(8 * 512 + 255) / 256, 256, 0, stream>>>(TMP, POOLED);
    final_k<<<(8 * 96 + 255) / 256, 256, 0, stream>>>(POOLED, out_w, out_b, (float*)d_out);
}
